// TrSFRNNAgent_77060303225061
// MI455X (gfx1250) — compile-verified, matrix operand order checked by reading, not run on hardware
//
#include <hip/hip_runtime.h>

// ---------------------------------------------------------------------------
// Fused TrSF RNN agent step for MI455X (gfx1250, wave32, WMMA).
// All heavy GEMMs use v_wmma_f32_16x16x32_bf16. Weights are pre-packed once
// into WMMA B-fragment order (bf16) in d_ws by pack_w.  Long K-loops use two
// independent accumulator chains to cover the WMMA->WMMA RAW hazard window.
// ---------------------------------------------------------------------------

#define OWN_D 96
#define ED    64
#define N_AL  9
#define N_EN  10
#define HID   256
#define EMB   256      // HEAD*ATTN == HEAD*ENT
#define FEAT  768
#define BROWS 16384

typedef __attribute__((ext_vector_type(16))) __bf16        v16bf;
typedef __attribute__((ext_vector_type(8)))  float         v8f;
typedef __attribute__((ext_vector_type(4)))  unsigned int  u32x4;

static __device__ __forceinline__ unsigned short f2bf(float f) {
    unsigned u = __builtin_bit_cast(unsigned, f);
    unsigned r = (u + 0x7FFFu + ((u >> 16) & 1u)) >> 16;   // RNE
    return (unsigned short)r;
}
static __device__ __forceinline__ unsigned pack2bf(float lo, float hi) {
    return (unsigned)f2bf(lo) | ((unsigned)f2bf(hi) << 16);
}
static __device__ __forceinline__ float bflo(unsigned u) {
    return __builtin_bit_cast(float, u << 16);
}
static __device__ __forceinline__ float bfhi(unsigned u) {
    return __builtin_bit_cast(float, u & 0xFFFF0000u);
}

// A-fragment (16xK tile, bf16 row-major in LDS, leading dim ldA).
// Per ISA 16-bit A layout: lanes 0-15 hold M=lane, K={0..7,16..23};
// lanes 16-31 hold M=lane-16, K={8..15,24..31}.  Two 16B LDS loads.
static __device__ __forceinline__ v16bf load_a(const unsigned short* sA, int ldA,
                                               int kt, int lane) {
    int m    = lane & 15;
    int koff = (lane >> 4) << 3;
    const unsigned short* p = sA + m * ldA + (kt << 5) + koff;
    union { u32x4 q[2]; v16bf v; } f;
    f.q[0] = *reinterpret_cast<const u32x4*>(p);
    f.q[1] = *reinterpret_cast<const u32x4*>(p + 16);
    return f.v;
}

// B-fragment: pre-packed so each lane loads 32 contiguous bytes (coalesced 1KB/frag).
static __device__ __forceinline__ v16bf load_b(const unsigned short* pW, int nKt,
                                               int nt, int kt, int lane) {
    const unsigned short* p = pW + ((((size_t)nt * nKt + kt) << 5) + lane) * 16;
    return *reinterpret_cast<const v16bf*>(p);
}

static __device__ __forceinline__ v8f wmma_acc(v16bf a, v16bf b, v8f c) {
    return __builtin_amdgcn_wmma_f32_16x16x32_bf16(false, a, false, b,
                                                   (short)0, c, false, false);
}

// Dual-chain GEMM over nKt K-fragments (nKt even): two independent WMMA
// dependency chains interleave so the compiler can fill the WMMA->WMMA
// hazard slots with the other chain instead of v_nop padding.
static __device__ __forceinline__ v8f gemm_dual(const unsigned short* sA, int ldA,
                                                const unsigned short* pW, int nKt,
                                                int nt, int lane,
                                                v8f c0, v8f c1, v8f* c1out) {
    for (int kt = 0; kt < nKt; kt += 2) {
        c0 = wmma_acc(load_a(sA, ldA, kt,     lane), load_b(pW, nKt, nt, kt,     lane), c0);
        c1 = wmma_acc(load_a(sA, ldA, kt + 1, lane), load_b(pW, nKt, nt, kt + 1, lane), c1);
    }
    *c1out = c1;
    return c0;
}

// ---------------------------------------------------------------------------
// Weight pre-pack: f32 [K x N] row-major  ->  bf16 WMMA B-fragment order.
// ---------------------------------------------------------------------------
__global__ void pack_w(const float* __restrict__ src, unsigned short* __restrict__ dst,
                       int K, int N) {
    int total = K * N;
    int nKt   = K >> 5;
    for (int idx = blockIdx.x * blockDim.x + threadIdx.x; idx < total;
         idx += gridDim.x * blockDim.x) {
        int frag = idx >> 9, rem = idx & 511;
        int lane = rem >> 4, i = rem & 15;
        int nt = frag / nKt, kt = frag - nt * nKt;
        int v = i >> 1, p = i & 1;
        int half = lane >> 4, nl = lane & 15;
        int kl = (v < 4 ? (v << 1) + p : 16 + ((v - 4) << 1) + p) + (half << 3);
        int k = (kt << 5) + kl, n = (nt << 4) + nl;
        dst[idx] = f2bf(src[(size_t)k * N + n]);
    }
}

// ---------------------------------------------------------------------------
// Single-query multi-head attention over one entity type, fused.
// ---------------------------------------------------------------------------
static __device__ __forceinline__ void attention(
    const float* __restrict__ feats, int nE,
    const unsigned short* pK, const float* bK,
    const unsigned short* pV, const float* bV,
    int xoff, int row0,
    unsigned short* sEnt, unsigned short* sV, float* sQ, float* sScore,
    unsigned short* sX, int tid, int wid, int lane)
{
    unsigned* sEnt32 = (unsigned*)sEnt;
    // ---- keys: nE chunks of 16 entity rows ----
    for (int ch = 0; ch < nE; ++ch) {
        for (int i = tid; i < 16 * (ED / 2); i += 256) {
            int r = i >> 5, c2 = i & 31;
            int e = ch * 16 + r;
            int b = e / nE, n = e - b * nE;
            const float2 s = *reinterpret_cast<const float2*>(
                feats + ((size_t)(row0 + b) * nE + n) * ED + c2 * 2);
            sEnt32[i] = pack2bf(s.x, s.y);
        }
        __syncthreads();
        for (int j = 0; j < 2; ++j) {
            int nt = wid * 2 + j;
            v8f acc; for (int q = 0; q < 8; ++q) acc[q] = 0.f;
            for (int kt = 0; kt < 2; ++kt)
                acc = wmma_acc(load_a(sEnt, ED, kt, lane), load_b(pK, 2, nt, kt, lane), acc);
            int nc = lane & 15, mh = (lane >> 4) << 3;
            float bb = bK[nt * 16 + nc];
            for (int r = 0; r < 8; ++r)
                sV[(size_t)(ch * 16 + r + mh) * EMB + nt * 16 + nc] = f2bf(acc[r] + bb);
        }
        __syncthreads();
    }
    // ---- energies: q . k / sqrt(64), vectorized b128 reads ----
    for (int idx = tid; idx < 64 * nE; idx += 256) {
        int b = idx / (4 * nE); int rem = idx - b * 4 * nE;
        int h = rem / nE;       int n = rem - h * nE;
        const float4* qp = reinterpret_cast<const float4*>(sQ + b * EMB + h * 64);
        const u32x4*  kp = reinterpret_cast<const u32x4*>(
            sV + (size_t)(b * nE + n) * EMB + h * 64);
        float s = 0.f;
        for (int it = 0; it < 8; ++it) {
            u32x4  kk = kp[it];
            float4 q0 = qp[it * 2], q1 = qp[it * 2 + 1];
            s += q0.x * bflo(kk.x) + q0.y * bfhi(kk.x)
               + q0.z * bflo(kk.y) + q0.w * bfhi(kk.y)
               + q1.x * bflo(kk.z) + q1.y * bfhi(kk.z)
               + q1.z * bflo(kk.w) + q1.w * bfhi(kk.w);
        }
        sScore[(b * 4 + h) * 12 + n] = s * 0.125f;
    }
    __syncthreads();
    // ---- softmax over n (length nE) ----
    if (tid < 64) {
        float* row = sScore + tid * 12;
        float mx = row[0];
        for (int n = 1; n < nE; ++n) mx = fmaxf(mx, row[n]);
        float sum = 0.f;
        for (int n = 0; n < nE; ++n) { float e = __expf(row[n] - mx); row[n] = e; sum += e; }
        float inv = 1.f / sum;
        for (int n = 0; n < nE; ++n) row[n] *= inv;
    }
    __syncthreads();
    // ---- values via WMMA ----
    for (int ch = 0; ch < nE; ++ch) {
        for (int i = tid; i < 16 * (ED / 2); i += 256) {
            int r = i >> 5, c2 = i & 31;
            int e = ch * 16 + r;
            int b = e / nE, n = e - b * nE;
            const float2 s = *reinterpret_cast<const float2*>(
                feats + ((size_t)(row0 + b) * nE + n) * ED + c2 * 2);
            sEnt32[i] = pack2bf(s.x, s.y);
        }
        __syncthreads();
        for (int j = 0; j < 2; ++j) {
            int nt = wid * 2 + j;
            v8f acc; for (int q = 0; q < 8; ++q) acc[q] = 0.f;
            for (int kt = 0; kt < 2; ++kt)
                acc = wmma_acc(load_a(sEnt, ED, kt, lane), load_b(pV, 2, nt, kt, lane), acc);
            int nc = lane & 15, mh = (lane >> 4) << 3;
            float bb = bV[nt * 16 + nc];
            for (int r = 0; r < 8; ++r)
                sV[(size_t)(ch * 16 + r + mh) * EMB + nt * 16 + nc] = f2bf(acc[r] + bb);
        }
        __syncthreads();
    }
    // ---- weighted sum -> x[:, xoff:xoff+256], packed dword stores ----
    {
        int b = tid >> 4, g = tid & 15;
        int h = g >> 2;                      // 16 cols never cross a head boundary
        float acc[16];
        for (int i = 0; i < 16; ++i) acc[i] = 0.f;
        for (int n = 0; n < nE; ++n) {
            float s = sScore[(b * 4 + h) * 12 + n];
            const u32x4* vr = reinterpret_cast<const u32x4*>(
                sV + (size_t)(b * nE + n) * EMB + g * 16);
            u32x4 v0 = vr[0], v1 = vr[1];
            acc[0]  += s * bflo(v0.x); acc[1]  += s * bfhi(v0.x);
            acc[2]  += s * bflo(v0.y); acc[3]  += s * bfhi(v0.y);
            acc[4]  += s * bflo(v0.z); acc[5]  += s * bfhi(v0.z);
            acc[6]  += s * bflo(v0.w); acc[7]  += s * bfhi(v0.w);
            acc[8]  += s * bflo(v1.x); acc[9]  += s * bfhi(v1.x);
            acc[10] += s * bflo(v1.y); acc[11] += s * bfhi(v1.y);
            acc[12] += s * bflo(v1.z); acc[13] += s * bfhi(v1.z);
            acc[14] += s * bflo(v1.w); acc[15] += s * bfhi(v1.w);
        }
        unsigned* dst = (unsigned*)(sX + b * FEAT + xoff + g * 16);
        for (int i = 0; i < 8; ++i)
            dst[i] = pack2bf(acc[2 * i], acc[2 * i + 1]);
    }
    __syncthreads();
}

// ---------------------------------------------------------------------------
// Main fused kernel: one block = 16 batch rows, 8 wave32.
// ---------------------------------------------------------------------------
__global__ void __launch_bounds__(256)
fused_agent(const float* __restrict__ own,  const float* __restrict__ ally,
            const float* __restrict__ enemy, const float* __restrict__ hin,
            const float* __restrict__ bq,   const float* __restrict__ bak,
            const float* __restrict__ bav,  const float* __restrict__ bek,
            const float* __restrict__ bev,  const float* __restrict__ bov,
            const float* __restrict__ b_ih, const float* __restrict__ b_hh,
            const unsigned short* __restrict__ ws, float* __restrict__ out)
{
    extern __shared__ __align__(16) char smem[];
    unsigned short* sOwn = (unsigned short*)smem;            // 16*96  bf16
    unsigned short* sH   = sOwn + 16 * OWN_D;                // 16*256 bf16
    unsigned short* sEnt = sH + 16 * HID;                    // 16*64  bf16
    unsigned short* sX   = sEnt + 16 * ED;                   // 16*768 bf16
    float* sQ     = (float*)(sX + 16 * FEAT);                // 16*256 f32
    float* sScore = sQ + 16 * EMB;                           // 16*4*12 f32
    float* sRZ    = sScore + 16 * 4 * 12;                    // 16*512 f32 (r|z)
    unsigned short* sV = (unsigned short*)(sRZ + 16 * 512);  // 160*256 bf16

    const int tid  = threadIdx.x;
    // wave-uniform SGPR wave id -> scalar branches, no EXEC save/restore
    const int wid  = __builtin_amdgcn_readfirstlane(tid >> 5);
    const int lane = tid & 31;
    const int row0 = blockIdx.x * 16;

    const unsigned short* pWq  = ws;
    const unsigned short* pWov = ws + 24576;
    const unsigned short* pWak = ws + 49152;
    const unsigned short* pWav = ws + 65536;
    const unsigned short* pWek = ws + 81920;
    const unsigned short* pWev = ws + 98304;
    const unsigned short* pWih = ws + 114688;
    const unsigned short* pWhh = ws + 704512;

    // stage own_obs and h as packed bf16 pairs
    {
        unsigned* d32 = (unsigned*)sOwn;
        for (int i = tid; i < 16 * (OWN_D / 2); i += 256) {
            int m = i / (OWN_D / 2), c2 = i - m * (OWN_D / 2);
            const float2 s = *reinterpret_cast<const float2*>(
                own + (size_t)(row0 + m) * OWN_D + c2 * 2);
            d32[i] = pack2bf(s.x, s.y);
        }
    }
    {
        unsigned* d32 = (unsigned*)sH;
        for (int i = tid; i < 16 * (HID / 2); i += 256) {
            int m = i >> 7, c2 = i & 127;
            const float2 s = *reinterpret_cast<const float2*>(
                hin + (size_t)(row0 + m) * HID + c2 * 2);
            d32[i] = pack2bf(s.x, s.y);
        }
    }
    __syncthreads();

    // phase 1: query (Wq) -> sQ (f32) and own_feature (Wov) -> sX[:,0:256]
    // waves 0-3: Q tiles, waves 4-7: OV tiles (wave-uniform branch)
    if (wid < 4) {
        for (int j = 0; j < 4; ++j) {
            int nt = wid * 4 + j;
            v8f acc; for (int q = 0; q < 8; ++q) acc[q] = 0.f;
            for (int kt = 0; kt < 3; ++kt)
                acc = wmma_acc(load_a(sOwn, OWN_D, kt, lane), load_b(pWq, 3, nt, kt, lane), acc);
            int nc = lane & 15, mh = (lane >> 4) << 3;
            float bb = bq[nt * 16 + nc];
            for (int r = 0; r < 8; ++r)
                sQ[(r + mh) * EMB + nt * 16 + nc] = acc[r] + bb;
        }
    } else {
        for (int j = 0; j < 4; ++j) {
            int nt = (wid - 4) * 4 + j;
            v8f acc; for (int q = 0; q < 8; ++q) acc[q] = 0.f;
            for (int kt = 0; kt < 3; ++kt)
                acc = wmma_acc(load_a(sOwn, OWN_D, kt, lane), load_b(pWov, 3, nt, kt, lane), acc);
            int nc = lane & 15, mh = (lane >> 4) << 3;
            float bb = bov[nt * 16 + nc];
            for (int r = 0; r < 8; ++r)
                sX[(r + mh) * FEAT + nt * 16 + nc] = f2bf(acc[r] + bb);
        }
    }
    __syncthreads();

    attention(ally,  N_AL, pWak, bak, pWav, bav, 256, row0,
              sEnt, sV, sQ, sScore, sX, tid, wid, lane);
    attention(enemy, N_EN, pWek, bek, pWev, bev, 512, row0,
              sEnt, sV, sQ, sScore, sX, tid, wid, lane);

    // GRU part A: r and z gates (output cols 0..511): sigmoid(gi + gh)
    for (int j = 0; j < 4; ++j) {
        int nt = wid + j * 8;                       // 0..31
        v8f c0, c1;
        for (int q = 0; q < 8; ++q) { c0[q] = 0.f; c1[q] = 0.f; }
        c0 = gemm_dual(sX, FEAT, pWih, 24, nt, lane, c0, c1, &c1);
        c0 = gemm_dual(sH, HID,  pWhh,  8, nt, lane, c0, c1, &c1);
        v8f acc = c0 + c1;
        int nc = lane & 15, mh = (lane >> 4) << 3;
        int col = nt * 16 + nc;
        float bb = b_ih[col] + b_hh[col];
        for (int r = 0; r < 8; ++r) {
            float g = acc[r] + bb;
            sRZ[(r + mh) * 512 + col] = 1.f / (1.f + __expf(-g));
        }
    }
    __syncthreads();

    // GRU part B: candidate gate (cols 512..767) + final output
    for (int j = 0; j < 2; ++j) {
        int nt = 32 + wid + j * 8;                  // 32..47
        v8f i0, i1, h0, h1;
        for (int q = 0; q < 8; ++q) { i0[q] = 0.f; i1[q] = 0.f; h0[q] = 0.f; h1[q] = 0.f; }
        i0 = gemm_dual(sX, FEAT, pWih, 24, nt, lane, i0, i1, &i1);
        h0 = gemm_dual(sH, HID,  pWhh,  8, nt, lane, h0, h1, &h1);
        v8f ai = i0 + i1;
        v8f ah = h0 + h1;
        int nc = lane & 15, mh = (lane >> 4) << 3;
        int col = nt * 16 + nc;
        int cout = col - 512;
        float bi = b_ih[col], bh = b_hh[col];
        for (int r = 0; r < 8; ++r) {
            int m = r + mh;
            float i_n = ai[r] + bi;
            float h_n = ah[r] + bh;
            float rr = sRZ[m * 512 + cout];
            float zz = sRZ[m * 512 + 256 + cout];
            float t = i_n + rr * h_n;
            float e = __expf(-2.f * t);
            float nn = (1.f - e) / (1.f + e);       // tanh
            float hp = hin[(size_t)(row0 + m) * HID + cout];
            out[(size_t)(row0 + m) * HID + cout] = (1.f - zz) * nn + zz * hp;
        }
    }
}

// ---------------------------------------------------------------------------
extern "C" void kernel_launch(void* const* d_in, const int* in_sizes, int n_in,
                              void* d_out, int out_size, void* d_ws, size_t ws_size,
                              hipStream_t stream) {
    const float* own   = (const float*)d_in[0];
    const float* ally  = (const float*)d_in[1];
    const float* enemy = (const float*)d_in[2];
    const float* hin   = (const float*)d_in[3];
    const float* Wq  = (const float*)d_in[4];  const float* bq  = (const float*)d_in[5];
    const float* Wak = (const float*)d_in[6];  const float* bak = (const float*)d_in[7];
    const float* Wav = (const float*)d_in[8];  const float* bav = (const float*)d_in[9];
    const float* Wek = (const float*)d_in[10]; const float* bek = (const float*)d_in[11];
    const float* Wev = (const float*)d_in[12]; const float* bev = (const float*)d_in[13];
    const float* Wov = (const float*)d_in[14]; const float* bov = (const float*)d_in[15];
    const float* Wih = (const float*)d_in[16]; const float* b_ih = (const float*)d_in[17];
    const float* Whh = (const float*)d_in[18]; const float* b_hh = (const float*)d_in[19];
    unsigned short* ws = (unsigned short*)d_ws;   // needs ~1.76 MB (901120 bf16)

    auto pk = [&](const float* src, size_t off, int K, int N) {
        int total = K * N;
        pack_w<<<dim3((total + 255) / 256), dim3(256), 0, stream>>>(src, ws + off, K, N);
    };
    pk(Wq,  0,      96,  256);
    pk(Wov, 24576,  96,  256);
    pk(Wak, 49152,  64,  256);
    pk(Wav, 65536,  64,  256);
    pk(Wek, 81920,  64,  256);
    pk(Wev, 98304,  64,  256);
    pk(Wih, 114688, 768, 768);
    pk(Whh, 704512, 256, 768);

    constexpr size_t SMEM =
        (size_t)(16 * OWN_D + 16 * HID + 16 * ED + 16 * FEAT) * 2  // bf16 staging
      + (size_t)(16 * EMB + 16 * 4 * 12 + 16 * 512) * 4            // f32 q/scores/rz
      + (size_t)(160 * EMB) * 2;                                   // bf16 K/V tile
    fused_agent<<<dim3(BROWS / 16), dim3(256), SMEM, stream>>>(
        own, ally, enemy, hin, bq, bak, bav, bek, bev, bov, b_ih, b_hh,
        ws, (float*)d_out);
}
